// MultilayerGRU_49005576848119
// MI455X (gfx1250) — matrix operand order repaired and checked by reading, compiled, not run on hardware
//
#include <hip/hip_runtime.h>
#include <hip/hip_bf16.h>

// GRU dims (fixed by the reference)
#define B_ 32
#define S_ 1024
#define I_ 256
#define H_ 512
#define O_ 256

typedef __attribute__((ext_vector_type(16))) __bf16 v16bf;
typedef __attribute__((ext_vector_type(8)))  __bf16 v8bf;
typedef __attribute__((ext_vector_type(8)))  float  v8f;

#define DEVINL static __device__ __forceinline__

DEVINL v8f wmma_bf16(v16bf a, v16bf b, v8f c) {
  // D = A(16x32) * B(32x16) + C, f32 accumulate
  return __builtin_amdgcn_wmma_f32_16x16x32_bf16(false, a, false, b, (short)0, c,
                                                 false, false);
}

// A-matrix fragment (16x32 bf16), row-major bf16 source.
// ISA 7.12.2: lanes 0-15 -> rows m0..m0+15, K = k0+{0..7, 16..23};
//             lanes 16-31 -> same rows,     K = k0+{8..15, 24..31}.
DEVINL v16bf frag_a_bf16(const __bf16* __restrict__ p, long long ld, int row0, int k0) {
  const int lane = threadIdx.x & 31;
  const __bf16* q = p + (size_t)(row0 + (lane & 15)) * (size_t)ld
                      + (size_t)(k0 + ((lane >> 4) << 3));
  v8bf lo = *(const v8bf*)q;          // 16B: K +0..7   (or +8..15)
  v8bf hi = *(const v8bf*)(q + 16);   // 16B: K +16..23 (or +24..31)
  return __builtin_shufflevector(lo, hi, 0, 1, 2, 3, 4, 5, 6, 7,
                                 8, 9, 10, 11, 12, 13, 14, 15);
}

// B-matrix fragment (32x16 bf16). We compute inp @ W^T, so B[k][n] = W[n][k]:
// load from W's row n. Lanes 0-15 -> col n0+lane, K = k0+{0..7, 16..23};
// lanes 16-31 -> col n0+lane-16, K = k0+{8..15, 24..31}.
DEVINL v16bf frag_b_bf16(const __bf16* __restrict__ p, int ld, int n0, int k0) {
  const int lane = threadIdx.x & 31;
  const __bf16* q = p + (size_t)(n0 + (lane & 15)) * (size_t)ld
                      + (size_t)(k0 + ((lane >> 4) << 3));
  v8bf lo = *(const v8bf*)q;
  v8bf hi = *(const v8bf*)(q + 16);
  return __builtin_shufflevector(lo, hi, 0, 1, 2, 3, 4, 5, 6, 7,
                                 8, 9, 10, 11, 12, 13, 14, 15);
}

DEVINL float sigmoidf_(float x) { return 1.0f / (1.0f + __expf(-x)); }

// ---------------------------------------------------------------------------
// Phase 1 (per layer, per timestep). One wave handles BOTH M-tiles (rows 0-31)
// of one 16-wide N slice, so each weight fragment is loaded once.
//   z    = sigmoid(inp@Wzx^T + bz + h@Wzh^T)         -> zbuf (f32)
//   rh   = sigmoid(inp@Wrx^T + br + h@Wrh^T) * h     -> rhbf (bf16, A-operand only)
//   gpart= inp@Wgx^T + bg                            -> gpart (f32)
// Grid: 32 blocks (N tiles), 32 threads (one wave).
// ---------------------------------------------------------------------------
__global__ void gru_phase1(const __bf16* __restrict__ inpbf, long long inp_ld, int din,
                           const __bf16* __restrict__ hbf,
                           const float* __restrict__ h,
                           const __bf16* __restrict__ Wzx, const __bf16* __restrict__ Wzh,
                           const __bf16* __restrict__ Wrx, const __bf16* __restrict__ Wrh,
                           const __bf16* __restrict__ Wgx,
                           const float* __restrict__ bz, const float* __restrict__ br,
                           const float* __restrict__ bg,
                           float* __restrict__ zbuf, __bf16* __restrict__ rhbf,
                           float* __restrict__ gpart) {
  const int n0 = blockIdx.x << 4;

  v8f az0 = {}, az1 = {}, ar0 = {}, ar1 = {}, ag0 = {}, ag1 = {};

  // Input projections: A fragments shared across 3 gate streams,
  // weight fragments shared across both M-tiles -> 6 WMMAs per k-step.
  for (int k0 = 0; k0 < din; k0 += 32) {
    v16bf a0 = frag_a_bf16(inpbf, inp_ld, 0, k0);
    v16bf a1 = frag_a_bf16(inpbf, inp_ld, 16, k0);
    v16bf wz = frag_b_bf16(Wzx, din, n0, k0);
    v16bf wr = frag_b_bf16(Wrx, din, n0, k0);
    v16bf wg = frag_b_bf16(Wgx, din, n0, k0);
    az0 = wmma_bf16(a0, wz, az0);  az1 = wmma_bf16(a1, wz, az1);
    ar0 = wmma_bf16(a0, wr, ar0);  ar1 = wmma_bf16(a1, wr, ar1);
    ag0 = wmma_bf16(a0, wg, ag0);  ag1 = wmma_bf16(a1, wg, ag1);
  }
  // Recurrent projections for z, r: 4 WMMAs per k-step.
  for (int k0 = 0; k0 < H_; k0 += 32) {
    v16bf a0 = frag_a_bf16(hbf, H_, 0, k0);
    v16bf a1 = frag_a_bf16(hbf, H_, 16, k0);
    v16bf wz = frag_b_bf16(Wzh, H_, n0, k0);
    v16bf wr = frag_b_bf16(Wrh, H_, n0, k0);
    az0 = wmma_bf16(a0, wz, az0);  az1 = wmma_bf16(a1, wz, az1);
    ar0 = wmma_bf16(a0, wr, ar0);  ar1 = wmma_bf16(a1, wr, ar1);
  }

  // C/D layout: lane l holds col n0+(l&15), rows m0+8*(l>>4)+i in acc[i].
  const int lane = threadIdx.x & 31;
  const int n  = n0 + (lane & 15);
  const int mh = (lane >> 4) << 3;
  const float vbz = bz[n], vbr = br[n], vbg = bg[n];
#pragma unroll
  for (int i = 0; i < 8; ++i) {
    // M-tile 0 (rows 0-15)
    {
      const int m = mh + i;
      const float hv = h[m * H_ + n];
      const float z = sigmoidf_(az0[i] + vbz);
      const float r = sigmoidf_(ar0[i] + vbr);
      zbuf[m * H_ + n]  = z;
      rhbf[m * H_ + n]  = (__bf16)(r * hv);
      gpart[m * H_ + n] = ag0[i] + vbg;
    }
    // M-tile 1 (rows 16-31)
    {
      const int m = 16 + mh + i;
      const float hv = h[m * H_ + n];
      const float z = sigmoidf_(az1[i] + vbz);
      const float r = sigmoidf_(ar1[i] + vbr);
      zbuf[m * H_ + n]  = z;
      rhbf[m * H_ + n]  = (__bf16)(r * hv);
      gpart[m * H_ + n] = ag1[i] + vbg;
    }
  }
}

// ---------------------------------------------------------------------------
// Phase 2 (per layer, per timestep):
//   g = tanh(gpart + rh@Wgh^T); h = z*h + (1-z)*g
//   writes fp32 state + bf16 mirror (+ optional bf16 copy as next layer input)
// Grid: 32 blocks (N tiles), 32 threads.
// ---------------------------------------------------------------------------
__global__ void gru_phase2(const __bf16* __restrict__ rhbf,
                           const __bf16* __restrict__ Wgh,
                           const float* __restrict__ gpart,
                           const float* __restrict__ zbuf,
                           float* __restrict__ h,
                           __bf16* __restrict__ hbf,
                           __bf16* __restrict__ inp_next_bf) {
  const int n0 = blockIdx.x << 4;

  const int lane = threadIdx.x & 31;
  const int n  = n0 + (lane & 15);
  const int mh = (lane >> 4) << 3;

  v8f acc0, acc1;
#pragma unroll
  for (int i = 0; i < 8; ++i) {
    acc0[i] = gpart[(mh + i) * H_ + n];
    acc1[i] = gpart[(16 + mh + i) * H_ + n];
  }

  for (int k0 = 0; k0 < H_; k0 += 32) {
    v16bf a0 = frag_a_bf16(rhbf, H_, 0, k0);
    v16bf a1 = frag_a_bf16(rhbf, H_, 16, k0);
    v16bf w  = frag_b_bf16(Wgh, H_, n0, k0);
    acc0 = wmma_bf16(a0, w, acc0);
    acc1 = wmma_bf16(a1, w, acc1);
  }

#pragma unroll
  for (int i = 0; i < 8; ++i) {
#pragma unroll
    for (int mt = 0; mt < 2; ++mt) {
      const int m = mt * 16 + mh + i;
      const float g  = tanhf(mt ? acc1[i] : acc0[i]);
      const float z  = zbuf[m * H_ + n];
      const float hv = h[m * H_ + n];
      const float hn = z * hv + (1.0f - z) * g;
      h[m * H_ + n]   = hn;
      const __bf16 hb = (__bf16)hn;
      hbf[m * H_ + n] = hb;
      if (inp_next_bf) inp_next_bf[m * H_ + n] = hb;
    }
  }
}

// ---------------------------------------------------------------------------
// Output projection: y[:, t, :] = h1 @ Wy^T + by.  Grid: 16 blocks (N tiles).
// ---------------------------------------------------------------------------
__global__ void out_proj(const __bf16* __restrict__ hbf,
                         const __bf16* __restrict__ Wy,
                         const float* __restrict__ by,
                         float* __restrict__ yout, long long out_bstride) {
  const int n0 = blockIdx.x << 4;

  v8f acc0 = {}, acc1 = {};
  for (int k0 = 0; k0 < H_; k0 += 32) {
    v16bf a0 = frag_a_bf16(hbf, H_, 0, k0);
    v16bf a1 = frag_a_bf16(hbf, H_, 16, k0);
    v16bf w  = frag_b_bf16(Wy, H_, n0, k0);
    acc0 = wmma_bf16(a0, w, acc0);
    acc1 = wmma_bf16(a1, w, acc1);
  }

  const int lane = threadIdx.x & 31;
  const int n  = n0 + (lane & 15);
  const int mh = (lane >> 4) << 3;
  const float vb = by[n];
#pragma unroll
  for (int i = 0; i < 8; ++i) {
    yout[(size_t)(mh + i) * (size_t)out_bstride + n]      = acc0[i] + vb;
    yout[(size_t)(16 + mh + i) * (size_t)out_bstride + n] = acc1[i] + vb;
  }
}

// ---------------------------------------------------------------------------
// Setup / teardown helpers
// ---------------------------------------------------------------------------
__global__ void cvt_bf16_k(const float* __restrict__ src, __bf16* __restrict__ dst, int n) {
  int i = blockIdx.x * blockDim.x + threadIdx.x;
  const int stride = gridDim.x * blockDim.x;
  for (; i < n; i += stride) dst[i] = (__bf16)src[i];
}

// h0 is [B, L, H]; internal h is [L, B, H] (fp32 master + bf16 mirror).
__global__ void init_h_k(const float* __restrict__ h0, float* __restrict__ h,
                         __bf16* __restrict__ hbf) {
  const int idx = blockIdx.x * blockDim.x + threadIdx.x;
  if (idx >= 2 * B_ * H_) return;
  const int l = idx / (B_ * H_);
  const int r = idx % (B_ * H_);
  const int b = r / H_, j = r % H_;
  const float v = h0[(b * 2 + l) * H_ + j];
  h[idx]   = v;
  hbf[idx] = (__bf16)v;
}

__global__ void final_h_k(const float* __restrict__ h, float* __restrict__ out) {
  const int idx = blockIdx.x * blockDim.x + threadIdx.x;
  if (idx >= 2 * B_ * H_) return;
  const int l = idx / (B_ * H_);
  const int r = idx % (B_ * H_);
  const int b = r / H_, j = r % H_;
  out[(b * 2 + l) * H_ + j] = h[idx];
}

// ---------------------------------------------------------------------------
extern "C" void kernel_launch(void* const* d_in, const int* in_sizes, int n_in,
                              void* d_out, int out_size, void* d_ws, size_t ws_size,
                              hipStream_t stream) {
  (void)in_sizes; (void)n_in; (void)out_size; (void)ws_size;

  const float* x  = (const float*)d_in[0];
  const float* h0 = (const float*)d_in[1];

  // Weight matrices (conversion order): per layer Wzx,Wzh,Wrx,Wrh,Wgx,Wgh; then Wy.
  static const int wsz[13] = {
      H_ * I_, H_ * H_, H_ * I_, H_ * H_, H_ * I_, H_ * H_,   // layer 0
      H_ * H_, H_ * H_, H_ * H_, H_ * H_, H_ * H_, H_ * H_,   // layer 1
      O_ * H_                                                  // Wy
  };
  static const int srcIdx[13] = {2, 4, 5, 7, 8, 10, 11, 13, 14, 16, 17, 19, 20};

  // ---- workspace layout ----
  __bf16* wb = (__bf16*)d_ws;
  size_t off[14];
  off[0] = 0;
  for (int i = 0; i < 13; ++i) off[i + 1] = off[i] + (size_t)wsz[i];

  __bf16* xbf     = wb + off[13];                 // [B][S][I] bf16 (8.4M elems)
  __bf16* hbf     = xbf + (size_t)B_ * S_ * I_;   // [L][B][H]
  __bf16* rhbf    = hbf + 2 * B_ * H_;            // [B][H]
  __bf16* inp1bf  = rhbf + B_ * H_;               // [B][H]

  size_t fbase = ((size_t)((char*)(inp1bf + B_ * H_) - (char*)d_ws) + 255) & ~(size_t)255;
  float* h     = (float*)((char*)d_ws + fbase);   // [L][B][H] fp32 master
  float* zbuf  = h + 2 * B_ * H_;                 // [B][H]
  float* gpart = zbuf + B_ * H_;                  // [B][H]

  // ---- one-time conversions (per call; deterministic) ----
  for (int i = 0; i < 13; ++i)
    cvt_bf16_k<<<256, 256, 0, stream>>>((const float*)d_in[srcIdx[i]], wb + off[i], wsz[i]);
  cvt_bf16_k<<<2048, 256, 0, stream>>>(x, xbf, B_ * S_ * I_);
  init_h_k<<<(2 * B_ * H_ + 255) / 256, 256, 0, stream>>>(h0, h, hbf);

  float* yout       = (float*)d_out;                          // [B][S][O]
  float* hidden_out = yout + (size_t)B_ * S_ * O_;            // [B][L][H]

  const float* l0_bz = (const float*)d_in[3];
  const float* l0_br = (const float*)d_in[6];
  const float* l0_bg = (const float*)d_in[9];
  const float* l1_bz = (const float*)d_in[12];
  const float* l1_br = (const float*)d_in[15];
  const float* l1_bg = (const float*)d_in[18];
  const float* by    = (const float*)d_in[21];

  for (int t = 0; t < S_; ++t) {
    // ---- layer 0 (input rows: xbf[b, t, :], row stride S*I) ----
    gru_phase1<<<32, 32, 0, stream>>>(
        xbf + (size_t)t * I_, (long long)S_ * I_, I_, hbf, h,
        wb + off[0], wb + off[1], wb + off[2], wb + off[3], wb + off[4],
        l0_bz, l0_br, l0_bg, zbuf, rhbf, gpart);
    gru_phase2<<<32, 32, 0, stream>>>(rhbf, wb + off[5], gpart, zbuf, h, hbf, inp1bf);

    // ---- layer 1 ----
    gru_phase1<<<32, 32, 0, stream>>>(
        inp1bf, (long long)H_, H_, hbf + B_ * H_, h + B_ * H_,
        wb + off[6], wb + off[7], wb + off[8], wb + off[9], wb + off[10],
        l1_bz, l1_br, l1_bg, zbuf, rhbf, gpart);
    gru_phase2<<<32, 32, 0, stream>>>(rhbf, wb + off[11], gpart, zbuf,
                                      h + B_ * H_, hbf + B_ * H_, (__bf16*)nullptr);

    // ---- output projection ----
    out_proj<<<16, 32, 0, stream>>>(hbf + B_ * H_, wb + off[12], by,
                                    yout + (size_t)t * O_, (long long)S_ * O_);
  }

  final_h_k<<<(2 * B_ * H_ + 255) / 256, 256, 0, stream>>>(h, hidden_out);
}
